// QuantizedLMHead_78975858639391
// MI455X (gfx1250) — matrix operand order repaired and checked by reading, not compiled, not run on hardware
//
#include <hip/hip_runtime.h>
#include <hip/hip_bf16.h>
#include <stdint.h>

// Problem sizes (fixed by the reference)
#define NROWS 4096   // B*S
#define DDIM  2048
#define VDIM  32000

typedef __attribute__((ext_vector_type(2)))  float        v2f;
typedef __attribute__((ext_vector_type(8)))  float        v8f;
typedef __attribute__((ext_vector_type(8)))  __bf16       v8bf;
typedef __attribute__((ext_vector_type(16))) __bf16       v16bf;
typedef __attribute__((ext_vector_type(4)))  unsigned int v4u;
typedef __attribute__((ext_vector_type(4)))  int          v4i;
typedef __attribute__((ext_vector_type(8)))  int          v8i;

#if defined(__has_builtin)
#  if __has_builtin(__builtin_amdgcn_tensor_load_to_lds) && \
      __has_builtin(__builtin_amdgcn_s_wait_tensorcnt)
#    define HAVE_TDM 1
#  endif
#endif
#ifndef HAVE_TDM
#  define HAVE_TDM 0
#endif

static __device__ __forceinline__ unsigned short f32_to_bf16(float f) {
  union { float f; unsigned int u; } c; c.f = f;
  unsigned int u = c.u;
  unsigned int r = u + 0x7FFFu + ((u >> 16) & 1u);   // round-to-nearest-even
  return (unsigned short)(r >> 16);
}

// ---------------------------------------------------------------------------
// Kernel 1: s = x @ U (f32 WMMA 16x16x4), then per-element codebook
// quantization and scale by lambda_x*lambda_w, emitted as bf16 into Aq[N,D].
// Block = 128 threads (4 waves), block tile 64x64, each wave 32x32 (2x2 WMMA).
// ---------------------------------------------------------------------------
__global__ __launch_bounds__(128)
void gemm1_quant(const float* __restrict__ x, const float* __restrict__ U,
                 const float* __restrict__ lam_x, const float* __restrict__ lam_w,
                 const float* __restrict__ codebook,
                 unsigned short* __restrict__ Aq) {
  __shared__ __align__(16) float As[64][17];
  __shared__ __align__(16) float Bs[16][64];

  const int tid  = threadIdx.x;
  const int lane = tid & 31;
  const int wave = tid >> 5;
  const int wm   = wave >> 1;
  const int wn   = wave & 1;
  const int l16  = lane & 15;
  const bool hi  = lane >= 16;

  const int n0 = blockIdx.x * 64;
  const int m0 = blockIdx.y * 64;

  v8f acc[2][2] = {};

  for (int k0 = 0; k0 < DDIM; k0 += 16) {
    #pragma unroll
    for (int i = 0; i < 8; ++i) {
      int idx = tid + i * 128;
      int r = idx >> 4, c = idx & 15;
      As[r][c] = x[(size_t)(m0 + r) * DDIM + k0 + c];
    }
    #pragma unroll
    for (int i = 0; i < 8; ++i) {
      int idx = tid + i * 128;
      int kr = idx >> 6, c = idx & 63;
      Bs[kr][c] = U[(size_t)(k0 + kr) * DDIM + n0 + c];
    }
    __syncthreads();

    #pragma unroll
    for (int kk = 0; kk < 16; kk += 4) {
      const int kf = kk + (hi ? 2 : 0);
      v2f a[2], b[2];
      #pragma unroll
      for (int mt = 0; mt < 2; ++mt) {
        int r = wm * 32 + mt * 16 + l16;
        a[mt].x = As[r][kf];
        a[mt].y = As[r][kf + 1];
      }
      #pragma unroll
      for (int nt = 0; nt < 2; ++nt) {
        int c = wn * 32 + nt * 16 + l16;
        b[nt].x = Bs[kf][c];
        b[nt].y = Bs[kf + 1][c];
      }
      #pragma unroll
      for (int mt = 0; mt < 2; ++mt)
        #pragma unroll
        for (int nt = 0; nt < 2; ++nt)
          acc[mt][nt] = __builtin_amdgcn_wmma_f32_16x16x4_f32(
              false, a[mt], false, b[nt], (short)0, acc[mt][nt], false, false);
    }
    __syncthreads();
  }

  float cb[16];
  #pragma unroll
  for (int k = 0; k < 16; ++k) cb[k] = codebook[k];

  #pragma unroll
  for (int nt = 0; nt < 2; ++nt) {
    const int col = n0 + wn * 32 + nt * 16 + l16;
    const float lx  = lam_x[col];
    const float lw  = lam_w[col];
    const float slx = (fabsf(lx) < 1e-8f) ? 1e-8f : lx;
    const float cf  = lx * lw;
    #pragma unroll
    for (int mt = 0; mt < 2; ++mt) {
      const int rbase = m0 + wm * 32 + mt * 16 + (hi ? 8 : 0);
      #pragma unroll
      for (int j = 0; j < 8; ++j) {
        float z = acc[mt][nt][j] / slx;
        float best = cb[0];
        float bd   = fabsf(z - cb[0]);
        #pragma unroll
        for (int k = 1; k < 16; ++k) {
          float d = fabsf(z - cb[k]);
          if (d < bd) { bd = d; best = cb[k]; }
        }
        Aq[(size_t)(rbase + j) * DDIM + col] = f32_to_bf16(best * cf);
      }
    }
  }
}

// ---------------------------------------------------------------------------
// Kernel 1b: Z_w[D,V] f32 -> Bt[V,D] bf16 (transpose+convert, LDS tiled).
// ---------------------------------------------------------------------------
__global__ __launch_bounds__(256)
void cvt_zw_t(const float* __restrict__ Zw, unsigned short* __restrict__ Bt) {
  __shared__ __align__(16) unsigned short T[64 * 80];   // [vcol][krow], pad 80

  const int tid = threadIdx.x;
  const int v0  = blockIdx.x * 64;
  const int k0  = blockIdx.y * 64;

  #pragma unroll
  for (int i = 0; i < 4; ++i) {
    int idx = tid + i * 256;            // 0..1023 float4s
    int r   = idx >> 4;                 // k row 0..63
    int c4  = idx & 15;                 // float4 within row
    float4 v = *(const float4*)(Zw + (size_t)(k0 + r) * VDIM + v0 + c4 * 4);
    int vc = c4 * 4;
    T[(vc + 0) * 80 + r] = f32_to_bf16(v.x);
    T[(vc + 1) * 80 + r] = f32_to_bf16(v.y);
    T[(vc + 2) * 80 + r] = f32_to_bf16(v.z);
    T[(vc + 3) * 80 + r] = f32_to_bf16(v.w);
  }
  __syncthreads();
  #pragma unroll
  for (int i = 0; i < 2; ++i) {
    int idx = tid + i * 256;            // 0..511 uint4s
    int vr  = idx >> 3;                 // v row 0..63
    int kc  = (idx & 7) * 8;            // 8 bf16 chunk
    *(uint4*)(Bt + (size_t)(v0 + vr) * DDIM + k0 + kc) = *(const uint4*)&T[vr * 80 + kc];
  }
}

// ---------------------------------------------------------------------------
// Kernel 2: logits = Aq(bf16)[N,D] @ Bt(bf16)[V,D]^T, f32 accumulate.
// Block 256 threads (8 waves), block tile 256x128; each wave 64x64 = 4x4
// WMMA accumulators (16 v_wmma per K-step, 8 fragment b128-pairs -> 0.5
// ds-load per wmma). Tiles staged by the Tensor Data Mover, double buffered,
// descriptor-padded to a 40-bf16 LDS row stride.
// ---------------------------------------------------------------------------
#define LSTRIDE   40
#define KTILES    (DDIM / 32)
#define ABUFSH    (256 * LSTRIDE)
#define BBUFSH    (128 * LSTRIDE)

#if HAVE_TDM
static __device__ __forceinline__ void tdm_load_tile(unsigned lds_off,
                                                     const unsigned short* gptr,
                                                     unsigned tensor_dim1,
                                                     unsigned tile_rows) {
  unsigned long long ga = (unsigned long long)(size_t)(const void*)gptr;
  v4u g0;
  g0.x = 1u;                                     // count=1, user descriptor
  g0.y = lds_off;                                // LDS byte address
  g0.z = (unsigned)ga;                           // global_addr[31:0]
  g0.w = (unsigned)(ga >> 32) | (2u << 30);      // global_addr[56:32] | type=2
  v8i g1;
  g1[0] = (int)((1u << 16)                       // data_size = 2 bytes
              | (1u << 20)                       // pad_enable
              | (3u << 22)                       // pad_interval: 16 DWORDs (one row)
              | (3u << 25));                     // pad_amount: 4 DWORDs -> stride 40 bf16
  g1[1] = (int)((unsigned)(DDIM & 0xffffu) << 16);          // tensor_dim0 = D (lo16)
  g1[2] = (int)((tensor_dim1 & 0xffffu) << 16);             // dim0 hi=0 | tensor_dim1 lo
  g1[3] = (int)((tensor_dim1 >> 16) | (32u << 16));         // dim1 hi | tile_dim0 = 32
  g1[4] = (int)tile_rows;                                   // tile_dim1
  g1[5] = DDIM;                                             // tensor_dim0_stride = D
  g1[6] = 0;
  g1[7] = 0;
  v4i z4 = {0, 0, 0, 0};
#if __clang_major__ >= 23
  v8i z8 = {0, 0, 0, 0, 0, 0, 0, 0};
  __builtin_amdgcn_tensor_load_to_lds(g0, g1, z4, z4, z8, 0);
#else
  __builtin_amdgcn_tensor_load_to_lds(g0, g1, z4, z4, 0);
#endif
}
#endif

static __device__ __forceinline__ void stage_tile_256(unsigned short* dst,
                                                      const unsigned short* src,
                                                      int tid) {
  // 256 rows x 32 bf16: one full row per thread
  const uint4* g = (const uint4*)(src + (size_t)tid * DDIM);
  uint4 d0 = g[0], d1 = g[1], d2 = g[2], d3 = g[3];
  uint4* d = (uint4*)&dst[tid * LSTRIDE];
  d[0] = d0; d[1] = d1; d[2] = d2; d[3] = d3;
}

static __device__ __forceinline__ void stage_tile_128(unsigned short* dst,
                                                      const unsigned short* src,
                                                      int tid) {
  // 128 rows x 32 bf16: half row per thread
  int r = tid >> 1, c = (tid & 1) * 16;
  const uint4* g = (const uint4*)(src + (size_t)r * DDIM + c);
  uint4 d0 = g[0];
  uint4 d1 = g[1];
  *(uint4*)&dst[r * LSTRIDE + c]     = d0;
  *(uint4*)&dst[r * LSTRIDE + c + 8] = d1;
}

__global__ __launch_bounds__(256)
void gemm2_bf16(const unsigned short* __restrict__ Aq,
                const unsigned short* __restrict__ Bt,
                float* __restrict__ out) {
  __shared__ __align__(16) unsigned short As2[2][ABUFSH];
  __shared__ __align__(16) unsigned short Bs2[2][BBUFSH];

  const int tid  = threadIdx.x;
  const int lane = tid & 31;
  const int wave = tid >> 5;       // 0..7
  const int wm   = wave & 3;       // 64-row strip
  const int wn   = wave >> 2;      // 64-col strip
  const int l16  = lane & 15;
  const bool hi  = lane >= 16;

  const int n0 = blockIdx.x * 128;
  const int m0 = blockIdx.y * 256;

  v8f acc[4][4] = {};

  const unsigned short* gA = Aq + (size_t)m0 * DDIM;   // +kt*32 per tile
  const unsigned short* gB = Bt + (size_t)n0 * DDIM;

#if HAVE_TDM
  const unsigned ldsA = (unsigned)(size_t)(void*)&As2[0][0];
  const unsigned ldsB = (unsigned)(size_t)(void*)&Bs2[0][0];
  if (wave == 0) {
    tdm_load_tile(ldsA, gA, (unsigned)NROWS, 256u);
    tdm_load_tile(ldsB, gB, (unsigned)VDIM, 128u);
  }
#else
  stage_tile_256(As2[0], gA, tid);
  stage_tile_128(Bs2[0], gB, tid);
#endif

  for (int kt = 0; kt < KTILES; ++kt) {
    const int cur = kt & 1;
#if HAVE_TDM
    if (wave == 0) {
      if (kt + 1 < KTILES) {
        tdm_load_tile(ldsA + (unsigned)((cur ^ 1) * ABUFSH * 2), gA + (kt + 1) * 32,
                      (unsigned)NROWS, 256u);
        tdm_load_tile(ldsB + (unsigned)((cur ^ 1) * BBUFSH * 2), gB + (kt + 1) * 32,
                      (unsigned)VDIM, 128u);
        __builtin_amdgcn_s_wait_tensorcnt(2);   // current tile's 2 loads done
      } else {
        __builtin_amdgcn_s_wait_tensorcnt(0);   // final tile fully resident
      }
    }
    asm volatile("" ::: "memory");   // TDM writes LDS behind the compiler's back
#else
    if (kt + 1 < KTILES) {
      stage_tile_256(As2[cur ^ 1], gA + (kt + 1) * 32, tid);
      stage_tile_128(Bs2[cur ^ 1], gB + (kt + 1) * 32, tid);
    }
#endif
    __syncthreads();

    const unsigned short* Ab = As2[cur];
    const unsigned short* Bb = Bs2[cur];

    // A frags: lo lanes K{0..7,16..23}, hi lanes K{8..15,24..31}
    v16bf a[4];
    #pragma unroll
    for (int mt = 0; mt < 4; ++mt) {
      const unsigned short* base = Ab + (wm * 64 + mt * 16 + l16) * LSTRIDE;
      v8bf p = *(const v8bf*)(base + (hi ? 8 : 0));
      v8bf q = *(const v8bf*)(base + (hi ? 24 : 16));
      a[mt] = __builtin_shufflevector(p, q, 0,1,2,3,4,5,6,7,8,9,10,11,12,13,14,15);
    }
    // B frags: lo lanes K{0..7,16..23}, hi lanes K{8..15,24..31}
    #pragma unroll
    for (int nt = 0; nt < 4; ++nt) {
      const unsigned short* base = Bb + (wn * 64 + nt * 16 + l16) * LSTRIDE;
      v8bf p = *(const v8bf*)(base + (hi ? 8 : 0));
      v8bf q = *(const v8bf*)(base + (hi ? 24 : 16));
      v16bf bf = __builtin_shufflevector(p, q, 0,1,2,3,4,5,6,7,8,9,10,11,12,13,14,15);
      #pragma unroll
      for (int mt = 0; mt < 4; ++mt)
        acc[mt][nt] = __builtin_amdgcn_wmma_f32_16x16x32_bf16(
            false, a[mt], false, bf, (short)0, acc[mt][nt], false, false);
    }
    __syncthreads();
  }

  #pragma unroll
  for (int mt = 0; mt < 4; ++mt) {
    const int rbase = m0 + wm * 64 + mt * 16 + (hi ? 8 : 0);
    #pragma unroll
    for (int nt = 0; nt < 4; ++nt) {
      const int col = n0 + wn * 64 + nt * 16 + l16;
      #pragma unroll
      for (int j = 0; j < 8; ++j)
        out[(size_t)(rbase + j) * VDIM + col] = acc[mt][nt][j];
    }
  }
}

// ---------------------------------------------------------------------------
extern "C" void kernel_launch(void* const* d_in, const int* in_sizes, int n_in,
                              void* d_out, int out_size, void* d_ws, size_t ws_size,
                              hipStream_t stream) {
  const float* x   = (const float*)d_in[0];
  const float* U   = (const float*)d_in[1];
  const float* lx  = (const float*)d_in[2];
  const float* lw  = (const float*)d_in[3];
  const float* Zw  = (const float*)d_in[4];
  const float* cb  = (const float*)d_in[5];
  float* out = (float*)d_out;

  unsigned short* Aq = (unsigned short*)d_ws;                         // 16.78 MB
  unsigned short* Bt = (unsigned short*)((char*)d_ws +
                        (size_t)NROWS * DDIM * sizeof(unsigned short)); // 131 MB

  gemm1_quant<<<dim3(DDIM / 64, NROWS / 64), 128, 0, stream>>>(x, U, lx, lw, cb, Aq);
  cvt_zw_t<<<dim3(VDIM / 64, DDIM / 64), 256, 0, stream>>>(Zw, Bt);
  gemm2_bf16<<<dim3(VDIM / 128, NROWS / 256), 256, 0, stream>>>(Aq, Bt, out);
}
